// DoubleCVAELSTM_14963666059830
// MI455X (gfx1250) — matrix operand order repaired and checked by reading, not compiled, not run on hardware
//
#include <hip/hip_runtime.h>
#include <hip/hip_bf16.h>

typedef __attribute__((ext_vector_type(16))) __bf16 v16bf;
typedef __attribute__((ext_vector_type(8)))  __bf16 v8bf;
typedef __attribute__((ext_vector_type(8)))  float  v8f;
typedef __attribute__((ext_vector_type(4)))  float  v4f;

#define WPB 4            // waves per block for matrix kernels
#define TPB (WPB * 32)

// ---------------- problem constants (shapes fixed by setup_inputs) ----------
#define BB   4096
#define TO   8
#define TP   12
#define HH   512
#define ZZ   64
#define HZ   576
#define G4H  (4 * HH)    // 2048
#define G4HZ (4 * HZ)    // 2304

// ---------------------------- WMMA helpers ----------------------------------
__device__ __forceinline__ v8f wmma_bf16(v16bf a, v16bf b, v8f c) {
  return __builtin_amdgcn_wmma_f32_16x16x32_bf16(false, a, false, b, (short)0, c,
                                                 false, false);
}

// A fragment, 16x32 bf16 (ISA 7.12.2): lane<16 -> row m, K in {ka..ka+7, ka+16..ka+23}
__device__ __forceinline__ v16bf ld_afrag(const __bf16* p) {
  v8bf lo = *(const v8bf*)p;
  v8bf hi = *(const v8bf*)(p + 16);
  v16bf r;
#pragma unroll
  for (int i = 0; i < 8; ++i) { r[i] = lo[i]; r[i + 8] = hi[i]; }
  return r;
}

__device__ __forceinline__ v16bf ld_afrag(const float* p) {
  const v4f* q = (const v4f*)p;
  v4f a = q[0], b = q[1], c = q[4], d = q[5];
  v16bf r;
#pragma unroll
  for (int i = 0; i < 4; ++i) {
    r[i]      = (__bf16)a[i];
    r[i + 4]  = (__bf16)b[i];
    r[i + 8]  = (__bf16)c[i];
    r[i + 12] = (__bf16)d[i];
  }
  return r;
}

// B fragment, 32x16 bf16: lane<16 -> col n, K in {kb..kb+7, kb+16..kb+23} (same as A)
__device__ __forceinline__ v16bf ld_bfrag(const __bf16* p) {
  return ld_afrag(p);
}

__device__ __forceinline__ float sigf(float x) { return 1.f / (1.f + __expf(-x)); }
__device__ __forceinline__ float tanh_fast(float x) {
  float e = __expf(-2.f * fabsf(x));
  float r = (1.f - e) / (1.f + e);
  return x < 0.f ? -r : r;
}

// ---------------- async global -> LDS copy (CDNA5, ASYNCcnt-tracked) --------
__device__ __forceinline__ void async_copy_b128(unsigned lds_byte_addr,
                                                const void* gsrc) {
  asm volatile("global_load_async_to_lds_b128 %0, %1, off"
               :: "v"(lds_byte_addr),
                  "v"((unsigned long long)(uintptr_t)gsrc)
               : "memory");
}
__device__ __forceinline__ void wait_async_le2() {
  asm volatile("s_wait_asynccnt 2" ::: "memory");
}
__device__ __forceinline__ void wait_async_0() {
  asm volatile("s_wait_asynccnt 0" ::: "memory");
}

// ---------------------------- prep kernels ----------------------------------
__global__ void k_cvt_bf16(const float* __restrict__ src, __bf16* __restrict__ dst, long n) {
  long i = (long)blockIdx.x * blockDim.x + threadIdx.x;
  if (i < n) dst[i] = (__bf16)src[i];
}

// src (R x C) row-major  ->  dst (C x R) row-major (i.e. transposed), bf16
__global__ void k_cvt_bf16_t(const float* __restrict__ src, __bf16* __restrict__ dst,
                             int R, int C) {
  long i = (long)blockIdx.x * blockDim.x + threadIdx.x;
  if (i >= (long)R * C) return;
  int r = (int)(i / C), c = (int)(i % C);
  dst[(long)c * R + r] = (__bf16)src[(long)r * C + c];
}

__global__ void k_bias_sum(const float* __restrict__ a, const float* __restrict__ b,
                           float* __restrict__ out, int n) {
  int i = blockIdx.x * blockDim.x + threadIdx.x;
  if (i < n) out[i] = a[i] + b[i];
}

// fused normalize + 2->H dense, bf16 out.  vel: (BT,2), out: (BT,H)
__global__ void k_embed(const float* __restrict__ vel, const float* __restrict__ mean,
                        const float* __restrict__ stdv, const float* __restrict__ encW,
                        const float* __restrict__ encb, __bf16* __restrict__ out, long BT) {
  long i = (long)blockIdx.x * blockDim.x + threadIdx.x;
  if (i >= BT * HH) return;
  long row = i / HH;
  int  h   = (int)(i % HH);
  float x0 = (vel[row * 2 + 0] - mean[0]) / stdv[0];
  float x1 = (vel[row * 2 + 1] - mean[1]) / stdv[1];
  out[i] = (__bf16)(x0 * encW[h] + x1 * encW[HH + h] + encb[h]);
}

// --------------------------- generic WMMA GEMM ------------------------------
// C(MxN) = A(MxK) @ W(NxK)^T + bias ; per-wave tile 32x64
template <typename AT>
__global__ void k_gemm(const AT* __restrict__ A, int K,
                       const __bf16* __restrict__ W, const float* __restrict__ bias,
                       float* __restrict__ outF, __bf16* __restrict__ outB,
                       int M, int N, int relu) {
  const int lane = threadIdx.x & 31;
  const int wid  = threadIdx.x >> 5;
  const int tiles_m = M >> 5;
  const int tiles_n = N >> 6;
  const int tile = blockIdx.x * WPB + wid;
  if (tile >= tiles_m * tiles_n) return;
  const int m0 = (tile % tiles_m) << 5;
  const int n0 = (tile / tiles_m) << 6;

  v8f acc[2][4];
#pragma unroll
  for (int s = 0; s < 2; ++s)
#pragma unroll
    for (int j = 0; j < 4; ++j)
#pragma unroll
      for (int e = 0; e < 8; ++e) acc[s][j][e] = 0.f;

  const int r  = lane & 15;
  const int ka = (lane >> 4) << 3;   // A K sub-offset
  const int kb = (lane >> 4) << 3;   // B K sub-offset

  const AT* a0 = A + (long)(m0 + r) * K + ka;
  const AT* a1 = a0 + (long)16 * K;
  const __bf16* w0 = W + (long)(n0 + r) * K + kb;

  for (int k = 0; k < K; k += 32) {
    v16bf af0 = ld_afrag(a0 + k);
    v16bf af1 = ld_afrag(a1 + k);
#pragma unroll
    for (int j = 0; j < 4; ++j) {
      v16bf bf = ld_bfrag(w0 + (long)(j * 16) * K + k);
      acc[0][j] = wmma_bf16(af0, bf, acc[0][j]);
      acc[1][j] = wmma_bf16(af1, bf, acc[1][j]);
    }
  }

  const int cnb   = n0 + r;
  const int rbase = m0 + ((lane >> 4) << 3);
#pragma unroll
  for (int j = 0; j < 4; ++j) {
    const int cn = cnb + j * 16;
    const float bv = bias ? bias[cn] : 0.f;
#pragma unroll
    for (int s = 0; s < 2; ++s) {
      const int rm = rbase + s * 16;
#pragma unroll
      for (int e = 0; e < 8; ++e) {
        float v = acc[s][j][e] + bv;
        if (relu && v < 0.f) v = 0.f;
        const long idx = (long)(rm + e) * N + cn;
        if (outF) outF[idx] = v;
        if (outB) outB[idx] = (__bf16)v;
      }
    }
  }
}

// --------------------------- fused LSTM step --------------------------------
// gates = x @ Wih^T + h_prev @ Whh^T + bias ; cell update fused in epilogue.
// Per-wave: 32 batch rows x 16 hidden cols, all 4 gates (8 fp32 accumulators).
// The 4 waves of a block share the same 16 hidden cols -> weights for the
// current K-chunk (4 gates x 16 cols x 32 K = 4KB) are staged in LDS once per
// block via async global->LDS copies, double-buffered against the WMMAs.
__global__ void __launch_bounds__(TPB)
k_lstm_step(const __bf16* __restrict__ x, long xs,
            const __bf16* hprev, long hs,
            const __bf16* __restrict__ Wih,
            const __bf16* __restrict__ Whh,
            const float* __restrict__ bias,
            float* c,                    // (M x Hd) fp32, in/out
            __bf16* hout, long hos,      // bf16 h for next GEMMs
            float* houtf, long hfs,      // optional fp32 h
            int M, int Hd, int first) {
  __shared__ __align__(32) __bf16 smem[2][2048];   // 2 x 4KB weight chunks

  const int lane = threadIdx.x & 31;
  const int wid  = threadIdx.x >> 5;
  const int t    = threadIdx.x;
  const int tiles_m = M >> 5;                      // multiple of WPB
  const int tile = blockIdx.x * WPB + wid;
  const int m0 = (tile % tiles_m) << 5;
  const int n0 = (tile / tiles_m) << 4;            // same for all 4 waves

  v8f acc[4][2];
#pragma unroll
  for (int g = 0; g < 4; ++g)
#pragma unroll
    for (int s = 0; s < 2; ++s)
#pragma unroll
      for (int e = 0; e < 8; ++e) acc[g][s][e] = 0.f;

  const int r  = lane & 15;
  const int ka = (lane >> 4) << 3;

  const __bf16* xa0 = x + (long)(m0 + r) * xs + ka;
  const __bf16* xa1 = xa0 + (long)16 * xs;
  const __bf16* ha0 = first ? nullptr : hprev + (long)(m0 + r) * hs + ka;
  const __bf16* ha1 = first ? nullptr : ha0 + (long)16 * hs;

  const unsigned lds0 = (unsigned)(uintptr_t)&smem[0][0];
  const unsigned lds1 = (unsigned)(uintptr_t)&smem[1][0];

  const int nk = Hd >> 5;
  const int total = first ? nk : (nk << 1);

  // producer: copy 4KB chunk (gate-major rows of 64B) for K-chunk kk of Wbase
  auto issue_chunk = [&](const __bf16* Wbase, int kk, unsigned lds_base) {
#pragma unroll
    for (int rep = 0; rep < 2; ++rep) {
      int idx = rep * 128 + t;                 // 256 x 16B segments
      int row = idx >> 2;                      // 0..63 = g*16 + col
      int g = row >> 4, cc = row & 15;
      const __bf16* src = Wbase + (long)(g * Hd + n0 + cc) * Hd + kk + (idx & 3) * 8;
      async_copy_b128(lds_base + (unsigned)idx * 16u, src);
    }
  };

  issue_chunk(Wih, 0, lds0);
  for (int q = 0; q < total; ++q) {
    const int qn = q + 1;
    if (qn < total) {
      const __bf16* Wb = (qn < nk) ? Wih : Whh;
      const int kk = ((qn < nk) ? qn : qn - nk) << 5;
      issue_chunk(Wb, kk, (qn & 1) ? lds1 : lds0);
      wait_async_le2();   // chunk q complete (in-order), chunk q+1 in flight
    } else {
      wait_async_0();
    }
    __syncthreads();

    const int kk = ((q < nk) ? q : q - nk) << 5;
    v16bf af0, af1;
    if (q < nk) { af0 = ld_afrag(xa0 + kk); af1 = ld_afrag(xa1 + kk); }
    else        { af0 = ld_afrag(ha0 + kk); af1 = ld_afrag(ha1 + kk); }
    const int qb = q & 1;
#pragma unroll
    for (int g = 0; g < 4; ++g) {
      v16bf bfg = ld_afrag(&smem[qb][((g * 16 + r) << 5) + ((lane >> 4) << 3)]);
      acc[g][0] = wmma_bf16(af0, bfg, acc[g][0]);
      acc[g][1] = wmma_bf16(af1, bfg, acc[g][1]);
    }
    __syncthreads();   // all waves done reading before buffer is rewritten
  }

  // epilogue: fused cell update
  const int cn = n0 + r;
  const float bi = bias[0 * Hd + cn];
  const float bf = bias[1 * Hd + cn];
  const float bg = bias[2 * Hd + cn];
  const float bo = bias[3 * Hd + cn];
  const int rbase = m0 + ((lane >> 4) << 3);
#pragma unroll
  for (int s = 0; s < 2; ++s) {
    const int rm = rbase + s * 16;
#pragma unroll
    for (int e = 0; e < 8; ++e) {
      const int row = rm + e;
      const long cidx = (long)row * Hd + cn;
      float iv = sigf(acc[0][s][e] + bi);
      float fv = sigf(acc[1][s][e] + bf);
      float gv = tanh_fast(acc[2][s][e] + bg);
      float ov = sigf(acc[3][s][e] + bo);
      float cp = first ? 0.f : c[cidx];
      float cnw = fv * cp + iv * gv;
      float hv  = ov * tanh_fast(cnw);
      c[cidx] = cnw;
      hout[(long)row * hos + cn] = (__bf16)hv;
      if (houtf) houtf[(long)row * hfs + cn] = hv;
    }
  }
}

// --------------------------- small VALU kernels -----------------------------
__global__ void k_mean12(const float* __restrict__ pf2, float* __restrict__ out) {
  long i = (long)blockIdx.x * blockDim.x + threadIdx.x;
  if (i >= (long)BB * ZZ) return;
  long b = i / ZZ; int z = (int)(i % ZZ);
  float s = 0.f;
#pragma unroll
  for (int t = 0; t < TP; ++t) s += pf2[(b * TP + t) * ZZ + z];
  out[i] = s * (1.f / TP);
}

__global__ void k_concat2(const float* __restrict__ a, const float* __restrict__ b,
                          float* __restrict__ out) {
  long i = (long)blockIdx.x * blockDim.x + threadIdx.x;
  if (i >= (long)BB * HZ) return;
  long bb = i / HZ; int j = (int)(i % HZ);
  out[i] = (j < HH) ? a[bb * HH + j] : b[bb * ZZ + (j - HH)];
}

// reparameterize + build inner0 (fp32 into predlist[:,0,:] and bf16 for decoder x)
__global__ void k_noise_finish(const float* __restrict__ nz, const float* __restrict__ eps,
                               const float* __restrict__ innerInp,
                               float* __restrict__ mu_out, float* __restrict__ lv_out,
                               float* __restrict__ predlist, __bf16* __restrict__ inner0b) {
  long i = (long)blockIdx.x * blockDim.x + threadIdx.x;
  if (i >= (long)BB * HZ) return;
  long b = i / HZ; int j = (int)(i % HZ);
  float v;
  if (j < HH) {
    v = innerInp[b * HH + j];
  } else {
    int zi = j - HH;
    float mu = nz[b * 2 * ZZ + zi];
    float lv = nz[b * 2 * ZZ + ZZ + zi];
    mu_out[b * ZZ + zi] = mu;
    lv_out[b * ZZ + zi] = lv;
    v = eps[b * ZZ + zi] * __expf(0.5f * lv) + mu;
  }
  predlist[(b * TP) * (long)HZ + j] = v;
  inner0b[i] = (__bf16)v;
}

__global__ void k_dec_out(const float* __restrict__ predlist, const float* __restrict__ decW,
                          const float* __restrict__ decb, float* __restrict__ outVel) {
  long i = (long)blockIdx.x * blockDim.x + threadIdx.x;
  if (i >= (long)BB * TP * 2) return;
  long bt = i >> 1; int j = (int)(i & 1);
  const float* row = predlist + bt * HZ;
  float s = decb[j];
  for (int k = 0; k < HZ; ++k) s += row[k] * decW[k * 2 + j];
  outVel[i] = s;
}

__global__ void k_pred(const float* __restrict__ outVel, const float* __restrict__ mean,
                       const float* __restrict__ stdv, const float* __restrict__ obs,
                       float* __restrict__ pred) {
  long i = (long)blockIdx.x * blockDim.x + threadIdx.x;
  if (i >= (long)BB * 2) return;
  long b = i >> 1; int j = (int)(i & 1);
  float acc = obs[(b * TO + (TO - 1)) * 2 + j];
#pragma unroll
  for (int t = 0; t < TP; ++t) {
    acc += outVel[(b * TP + t) * 2 + j] * stdv[j] + mean[j];
    pred[(b * TP + t) * 2 + j] = acc;
  }
}

// ------------------------------- launcher -----------------------------------
static inline int cdiv(long a, long b) { return (int)((a + b - 1) / b); }

extern "C" void kernel_launch(void* const* d_in, const int* in_sizes, int n_in,
                              void* d_out, int out_size, void* d_ws, size_t ws_size,
                              hipStream_t stream) {
  const float* obs      = (const float*)d_in[0];
  const float* obs_vel  = (const float*)d_in[1];
  const float* pred_vel = (const float*)d_in[2];
  const float* meanp    = (const float*)d_in[3];
  const float* stdp     = (const float*)d_in[4];
  const float* eps      = (const float*)d_in[5];
  const float* enc_W    = (const float*)d_in[6];
  const float* enc_b    = (const float*)d_in[7];
  const float* dec_W    = (const float*)d_in[8];
  const float* dec_b    = (const float*)d_in[9];
  const float* amp_W    = (const float*)d_in[10];
  const float* amp_b    = (const float*)d_in[11];
  const float* cvae_W1  = (const float*)d_in[12];
  const float* cvae_b1  = (const float*)d_in[13];
  const float* cvae_W2  = (const float*)d_in[14];
  const float* cvae_b2  = (const float*)d_in[15];
  const float* noise_W1 = (const float*)d_in[16];
  const float* noise_b1 = (const float*)d_in[17];
  const float* noise_W2 = (const float*)d_in[18];
  const float* noise_b2 = (const float*)d_in[19];
  const float* l_Wih    = (const float*)d_in[20];
  const float* l_Whh    = (const float*)d_in[21];
  const float* l_bih    = (const float*)d_in[22];
  const float* l_bhh    = (const float*)d_in[23];
  const float* c_Wih    = (const float*)d_in[24];
  const float* c_Whh    = (const float*)d_in[25];
  const float* c_bih    = (const float*)d_in[26];
  const float* c_bhh    = (const float*)d_in[27];

  float* out_pred   = (float*)d_out;                               // B*TP*2
  float* out_mu     = out_pred + (long)BB * TP * 2;                // B*Z
  float* out_logvar = out_mu + (long)BB * ZZ;                      // B*Z
  float* out_outVel = out_logvar + (long)BB * ZZ;                  // B*TP*2

  // -------- workspace layout --------
  char* base = (char*)d_ws;
  size_t off = 0;
  auto alloc = [&](size_t bytes) -> void* {
    void* p = base + off;
    off = (off + bytes + 255) & ~(size_t)255;
    return p;
  };
  __bf16* Wihb   = (__bf16*)alloc((size_t)2 * G4H * HH * 2);
  __bf16* Whhb   = (__bf16*)alloc((size_t)2 * G4H * HH * 2);
  __bf16* cWihb  = (__bf16*)alloc((size_t)2 * G4HZ * HZ * 2);
  __bf16* cWhhb  = (__bf16*)alloc((size_t)2 * G4HZ * HZ * 2);
  __bf16* ampWt  = (__bf16*)alloc((size_t)HZ * HH * 2);
  __bf16* cvW1t  = (__bf16*)alloc((size_t)256 * HH * 2);
  __bf16* cvW2t  = (__bf16*)alloc((size_t)ZZ * 256 * 2);
  __bf16* nW1t   = (__bf16*)alloc((size_t)ZZ * HZ * 2);
  __bf16* nW2t   = (__bf16*)alloc((size_t)2 * ZZ * ZZ * 2);
  float*  bsum1  = (float*)alloc((size_t)2 * G4H * 4);
  float*  bsumc  = (float*)alloc((size_t)2 * G4HZ * 4);
  __bf16* embed  = (__bf16*)alloc((size_t)BB * TO * HH * 2);
  __bf16* pf0    = (__bf16*)alloc((size_t)BB * TP * HH * 2);
  __bf16* pf1    = (__bf16*)alloc((size_t)BB * TP * 256 * 2);
  float*  pf2    = (float*)alloc((size_t)BB * TP * ZZ * 4);
  float*  predF  = (float*)alloc((size_t)BB * ZZ * 4);
  __bf16* hseq0  = (__bf16*)alloc((size_t)BB * TO * HH * 2);
  __bf16* hseq1  = (__bf16*)alloc((size_t)BB * TO * HH * 2);
  float*  c_enc0 = (float*)alloc((size_t)BB * HH * 4);
  float*  c_enc1 = (float*)alloc((size_t)BB * HH * 4);
  float*  h0fin  = (float*)alloc((size_t)BB * HH * 4);
  float*  h1fin  = (float*)alloc((size_t)BB * HH * 4);   // == innerInp
  float*  noiseI = (float*)alloc((size_t)BB * HZ * 4);
  float*  n1     = (float*)alloc((size_t)BB * ZZ * 4);
  float*  nzb    = (float*)alloc((size_t)BB * 2 * ZZ * 4);
  __bf16* inner0 = (__bf16*)alloc((size_t)BB * HZ * 2);
  __bf16* h0buf0 = (__bf16*)alloc((size_t)BB * HZ * 2);
  __bf16* h0buf1 = (__bf16*)alloc((size_t)BB * HZ * 2);
  __bf16* h1buf0 = (__bf16*)alloc((size_t)BB * HZ * 2);
  __bf16* h1buf1 = (__bf16*)alloc((size_t)BB * HZ * 2);
  float*  c0d    = (float*)alloc((size_t)BB * HZ * 4);
  float*  c1d    = (float*)alloc((size_t)BB * HZ * 4);
  float*  plist  = (float*)alloc((size_t)BB * TP * HZ * 4);
  __bf16* h0bufs[2] = {h0buf0, h0buf1};
  __bf16* h1bufs[2] = {h1buf0, h1buf1};
  (void)ws_size; (void)in_sizes; (void)n_in; (void)out_size;

  // -------- weight prep --------
  {
    long n;
    n = (long)2 * G4H * HH;
    k_cvt_bf16<<<cdiv(n, 256), 256, 0, stream>>>(l_Wih, Wihb, n);
    k_cvt_bf16<<<cdiv(n, 256), 256, 0, stream>>>(l_Whh, Whhb, n);
    n = (long)2 * G4HZ * HZ;
    k_cvt_bf16<<<cdiv(n, 256), 256, 0, stream>>>(c_Wih, cWihb, n);
    k_cvt_bf16<<<cdiv(n, 256), 256, 0, stream>>>(c_Whh, cWhhb, n);
    k_cvt_bf16_t<<<cdiv((long)HH * HZ, 256), 256, 0, stream>>>(amp_W, ampWt, HH, HZ);
    k_cvt_bf16_t<<<cdiv((long)HH * 256, 256), 256, 0, stream>>>(cvae_W1, cvW1t, HH, 256);
    k_cvt_bf16_t<<<cdiv((long)256 * ZZ, 256), 256, 0, stream>>>(cvae_W2, cvW2t, 256, ZZ);
    k_cvt_bf16_t<<<cdiv((long)HZ * ZZ, 256), 256, 0, stream>>>(noise_W1, nW1t, HZ, ZZ);
    k_cvt_bf16_t<<<cdiv((long)ZZ * 2 * ZZ, 256), 256, 0, stream>>>(noise_W2, nW2t, ZZ, 2 * ZZ);
    k_bias_sum<<<cdiv(2 * G4H, 256), 256, 0, stream>>>(l_bih, l_bhh, bsum1, 2 * G4H);
    k_bias_sum<<<cdiv(2 * G4HZ, 256), 256, 0, stream>>>(c_bih, c_bhh, bsumc, 2 * G4HZ);
  }

  // -------- embeddings --------
  k_embed<<<cdiv((long)BB * TO * HH, 256), 256, 0, stream>>>(obs_vel, meanp, stdp, enc_W,
                                                             enc_b, embed, (long)BB * TO);
  k_embed<<<cdiv((long)BB * TP * HH, 256), 256, 0, stream>>>(pred_vel, meanp, stdp, enc_W,
                                                             enc_b, pf0, (long)BB * TP);

  // -------- CVAE feature path --------
  {
    int M = BB * TP;
    int tiles = (M >> 5) * (256 >> 6);
    k_gemm<__bf16><<<cdiv(tiles, WPB), TPB, 0, stream>>>(pf0, HH, cvW1t, cvae_b1, nullptr,
                                                         pf1, M, 256, 1);
    tiles = (M >> 5) * (ZZ >> 6);
    k_gemm<__bf16><<<cdiv(tiles, WPB), TPB, 0, stream>>>(pf1, 256, cvW2t, cvae_b2, pf2,
                                                         nullptr, M, ZZ, 0);
    k_mean12<<<cdiv((long)BB * ZZ, 256), 256, 0, stream>>>(pf2, predF);
  }

  // -------- encoder LSTM (2 layers, 8 steps) --------
  {
    int tiles = (BB >> 5) * (HH >> 4);
    int grid = tiles / WPB;               // exact
    long sseq = (long)TO * HH;
    for (int t = 0; t < TO; ++t) {
      k_lstm_step<<<grid, TPB, 0, stream>>>(
          embed + (long)t * HH, sseq, (t ? hseq0 + (long)(t - 1) * HH : nullptr), sseq,
          Wihb, Whhb, bsum1, c_enc0, hseq0 + (long)t * HH, sseq,
          (t == TO - 1) ? h0fin : nullptr, HH, BB, HH, t == 0);
    }
    const __bf16* Wih1 = Wihb + (long)G4H * HH;
    const __bf16* Whh1 = Whhb + (long)G4H * HH;
    for (int t = 0; t < TO; ++t) {
      k_lstm_step<<<grid, TPB, 0, stream>>>(
          hseq0 + (long)t * HH, sseq, (t ? hseq1 + (long)(t - 1) * HH : nullptr), sseq,
          Wih1, Whh1, bsum1 + G4H, c_enc1, hseq1 + (long)t * HH, sseq,
          (t == TO - 1) ? h1fin : nullptr, HH, BB, HH, t == 0);
    }
  }

  // -------- amp projections (decoder initial states) --------
  {
    int tiles = (BB >> 5) * (HZ >> 6);
    int grid = cdiv(tiles, WPB);
    k_gemm<float><<<grid, TPB, 0, stream>>>(h0fin, HH, ampWt, amp_b, nullptr, h0buf0, BB, HZ, 0);
    k_gemm<float><<<grid, TPB, 0, stream>>>(h1fin, HH, ampWt, amp_b, nullptr, h1buf0, BB, HZ, 0);
    k_gemm<float><<<grid, TPB, 0, stream>>>(c_enc0, HH, ampWt, amp_b, c0d, nullptr, BB, HZ, 0);
    k_gemm<float><<<grid, TPB, 0, stream>>>(c_enc1, HH, ampWt, amp_b, c1d, nullptr, BB, HZ, 0);
  }

  // -------- noise path + reparameterize --------
  {
    k_concat2<<<cdiv((long)BB * HZ, 256), 256, 0, stream>>>(h1fin, predF, noiseI);
    int tiles = (BB >> 5) * (ZZ >> 6);
    k_gemm<float><<<cdiv(tiles, WPB), TPB, 0, stream>>>(noiseI, HZ, nW1t, noise_b1, n1,
                                                        nullptr, BB, ZZ, 1);
    tiles = (BB >> 5) * ((2 * ZZ) >> 6);
    k_gemm<float><<<cdiv(tiles, WPB), TPB, 0, stream>>>(n1, ZZ, nW2t, noise_b2, nzb,
                                                        nullptr, BB, 2 * ZZ, 0);
    k_noise_finish<<<cdiv((long)BB * HZ, 256), 256, 0, stream>>>(nzb, eps, h1fin, out_mu,
                                                                 out_logvar, plist, inner0);
  }

  // -------- decoder LSTM (2 layers, pred_len-1 = 11 steps) --------
  {
    int tiles = (BB >> 5) * (HZ >> 4);
    int grid = tiles / WPB;               // exact
    const __bf16* cWih1 = cWihb + (long)G4HZ * HZ;
    const __bf16* cWhh1 = cWhhb + (long)G4HZ * HZ;
    for (int s = 0; s < TP - 1; ++s) {
      int p = s & 1, q = p ^ 1;
      const __bf16* x0 = (s == 0) ? inner0 : h1bufs[p];
      k_lstm_step<<<grid, TPB, 0, stream>>>(x0, HZ, h0bufs[p], HZ, cWihb, cWhhb, bsumc,
                                            c0d, h0bufs[q], HZ, nullptr, 0, BB, HZ, 0);
      k_lstm_step<<<grid, TPB, 0, stream>>>(h0bufs[q], HZ, h1bufs[p], HZ, cWih1, cWhh1,
                                            bsumc + G4HZ, c1d, h1bufs[q], HZ,
                                            plist + (long)(s + 1) * HZ, (long)TP * HZ,
                                            BB, HZ, 0);
    }
  }

  // -------- decoder dense + cumsum outputs --------
  k_dec_out<<<cdiv((long)BB * TP * 2, 256), 256, 0, stream>>>(plist, dec_W, dec_b, out_outVel);
  k_pred<<<cdiv((long)BB * 2, 256), 256, 0, stream>>>(out_outVel, meanp, stdp, obs, out_pred);
}
